// quantized_conv_21569325760825
// MI455X (gfx1250) — matrix operand order repaired and checked by reading, not compiled, not run on hardware
//
#include <hip/hip_runtime.h>

typedef __attribute__((ext_vector_type(16))) _Float16     v16h;
typedef __attribute__((ext_vector_type(8)))  float        v8f;
typedef __attribute__((ext_vector_type(4)))  unsigned int v4u;
typedef __attribute__((ext_vector_type(8)))  unsigned int v8u;

// Problem constants (fixed by the reference)
static const int B_  = 32;
static const int C_  = 256;
static const int HW_ = 32;          // H == W == 32
static const int O_  = 256;
static const int KF_ = 2304;        // C*3*3 features
static const int L_  = 1024;        // 32*32 positions

// ---------------------------------------------------------------------------
// Kernel 0: init workspace scalars and the a_loss output element
// ---------------------------------------------------------------------------
__global__ void k_init(unsigned int* maxbuf, float* out, int out_size) {
  if (blockIdx.x == 0 && threadIdx.x == 0) {
    maxbuf[0] = 0u;                  // abs-max accumulator for x (bits of 0.0f)
    maxbuf[1] = 0u;                  // abs-max accumulator for w
    out[(long)out_size - 1] = 0.0f;  // a_loss = 0
  }
}

// ---------------------------------------------------------------------------
// Kernel 1: grid-stride abs-max reduction (atomicMax on float bit patterns)
// ---------------------------------------------------------------------------
__global__ __launch_bounds__(256) void k_absmax(const float* __restrict__ src,
                                                long n, unsigned int* dst) {
  float m = 0.0f;
  for (long i = (long)blockIdx.x * blockDim.x + threadIdx.x; i < n;
       i += (long)gridDim.x * blockDim.x)
    m = fmaxf(m, fabsf(src[i]));
  for (int off = 16; off > 0; off >>= 1)
    m = fmaxf(m, __shfl_down(m, off, 32));
  __shared__ float red[8];
  const int lane = threadIdx.x & 31;
  const int wv   = threadIdx.x >> 5;
  if (lane == 0) red[wv] = m;
  __syncthreads();
  if (threadIdx.x == 0) {
    float t = red[0];
    for (int i = 1; i < (int)(blockDim.x >> 5); ++i) t = fmaxf(t, red[i]);
    atomicMax(dst, __float_as_uint(t));
  }
}

// ---------------------------------------------------------------------------
// Kernel 2: quantize f32 -> integer-valued fp16 code k = round(v/s * levels)
// ---------------------------------------------------------------------------
__global__ __launch_bounds__(256) void k_quant(const float* __restrict__ src,
                                               _Float16* __restrict__ dst,
                                               long n,
                                               const unsigned int* __restrict__ maxbuf,
                                               int slot, float levels) {
  const float s = __uint_as_float(maxbuf[slot]) + 1e-12f;
  const float r = levels / s;
  const long  i = (long)blockIdx.x * blockDim.x + threadIdx.x;
  if (i < n) dst[i] = (_Float16)rintf(src[i] * r);
}

// ---------------------------------------------------------------------------
// Kernel 3: fused im2col + 9x chunked GEMM + per-chunk ADC + accumulate.
//   Block = 256 threads (8 waves), tile 64(o) x 128(l); wave = 16x64 (4 WMMA).
//   Per group (256 K): TDM loads 64x256 fp16 weight panel to LDS; B tiles
//   [32k x 128n] are gathered branchlessly and double-buffered in LDS.
// ---------------------------------------------------------------------------
template <bool UX, bool UW>
__global__ __launch_bounds__(256) void k_main(
    const float* __restrict__ x, const float* __restrict__ w,
    const _Float16* __restrict__ X16, const _Float16* __restrict__ W16,
    const unsigned int* __restrict__ maxbuf, float* __restrict__ out) {

  __shared__ v16h BsV[2][128 * 2];                     // 2 x 8KB B-tile buffers
  __shared__ __align__(32) _Float16 Apanel[64 * 256];  // 32KB weight panel

  const int tid   = threadIdx.x;
  const int lane  = tid & 31;
  const int wv    = tid >> 5;
  const int ofrag = wv & 3;        // 16-row o sub-tile
  const int nhalf = wv >> 2;       // 64-col half of the l tile

  const int bid = blockIdx.x;
  const int ob  = bid & 3;         // 4 x 64 o-blocks
  const int lb  = (bid >> 2) & 7;  // 8 x 128 l-blocks
  const int b   = bid >> 5;        // 32 batches

  const float sa      = __uint_as_float(maxbuf[0]) + 1e-12f;
  const float sw      = __uint_as_float(maxbuf[1]) + 1e-12f;
  const float rs_a    = 255.0f / sa;
  const float rs_w    = 15.0f  / sw;
  const float invstep = sa * sw * 0.999f / (255.0f * 8.0f * 225.0f) * 1000.0f;
  const float step    = 1e-3f;

  // staging role: thread -> (column n, k half)
  const int sn    = tid & 127;
  const int skh   = tid >> 7;
  const int l_s   = lb * 128 + sn;
  const int ys    = l_s >> 5;
  const int xs    = l_s & 31;
  const int basec = b * (C_ * HW_ * HW_);

  // wmma lane roles
  const int col  = lane & 15;
  const int hi   = lane >> 4;
  const int arow = ofrag * 16 + col;   // o row within the LDS panel
  const int oA   = ob * 64 + arow;     // global o row (fallback path)

  // branchless im2col gather of one 16-half column chunk into dstBuf
  auto stageB = [&](int kk, v16h* dstBuf) {
    const int f0 = kk + skh * 16;      // feature = c*9 + ky*3 + kx
    int c  = f0 / 9;
    int r  = f0 - c * 9;
    int ky = r / 3;
    int kx = r - ky * 3;
    v16h bv;
    #pragma unroll
    for (int j = 0; j < 16; ++j) {
      const int  iy  = ys + ky - 1;
      const int  ix  = xs + kx - 1;
      const bool inr = ((unsigned)iy < 32u) && ((unsigned)ix < 32u);
      const int  idx = basec + (c << 10) + ((inr ? iy : 0) << 5) + (inr ? ix : 0);
      _Float16 lv;
      if (UX) lv = X16[idx];
      else    lv = (_Float16)rintf(x[idx] * rs_a);
      bv[j] = inr ? lv : (_Float16)0.0f;
      ++kx;
      if (kx == 3) { kx = 0; ++ky; if (ky == 3) { ky = 0; ++c; } }
    }
    dstBuf[sn * 2 + skh] = bv;         // two aligned ds_store_b128
  };

  const v8f zero = {0.f, 0.f, 0.f, 0.f, 0.f, 0.f, 0.f, 0.f};
  v8f tot0 = zero, tot1 = zero, tot2 = zero, tot3 = zero;

  for (int g = 0; g < 9; ++g) {
    __syncthreads();   // prior readers of Apanel / BsV are done

    if (UW) {
      if (wv == 0) {
        // ---- TDM: load 64(o) x 256(k) fp16 tile of W16 into Apanel ----
        const unsigned long long ga =
            (unsigned long long)(const void*)W16 +
            ((unsigned long long)(ob * 64) * (unsigned)KF_ + (unsigned)(g * 256)) * 2ull;
        const unsigned ldsA = (unsigned)(unsigned long long)(const void*)&Apanel[0];
        v4u g0;
        g0[0] = 1u;                                   // count=1, load, no gather
        g0[1] = ldsA;                                 // lds_addr (bytes)
        g0[2] = (unsigned)ga;                         // global_addr[31:0]
        g0[3] = ((unsigned)(ga >> 32) & 0x01FFFFFFu)  // global_addr[56:32]
                | 0x80000000u;                        // type = 2 ("image")
        v8u g1;
        g1[0] = 1u << 16;                             // data_size = 2 bytes
        g1[1] = (unsigned)KF_ << 16;                  // tensor_dim0[15:0] @63:48
        g1[2] = ((unsigned)KF_ >> 16)                 // tensor_dim0[31:16]
                | (256u << 16);                       // tensor_dim1[15:0] @111:80
        g1[3] = 256u << 16;                           // tile_dim0 = 256 @127:112
        g1[4] = 64u;                                  // tile_dim1 = 64
        g1[5] = (unsigned)KF_;                        // tensor_dim0_stride = 2304
        g1[6] = 0u;
        g1[7] = 0u;
        asm volatile("tensor_load_to_lds %0, %1" :: "s"(g0), "s"(g1) : "memory");
        __builtin_amdgcn_s_wait_tensorcnt(0);
      }
    }

    // prologue: stage B tile for ks = 0
    stageB(g * 256, BsV[0]);
    __syncthreads();   // panel + BsV[0] ready

    v8f acc0 = zero, acc1 = zero, acc2 = zero, acc3 = zero;

    for (int ks = 0; ks < 8; ++ks) {
      // ---- A fragment: halves 0..15 <- K+16*hi+s ----
      v16h a;
      if (UW) {
        const _Float16* ap = Apanel + arow * 256 + ks * 32 + 16 * hi;
        union { v16h v; uint4 q[2]; } u;
        u.q[0] = *(const uint4*)(ap);
        u.q[1] = *(const uint4*)(ap + 8);
        a = u.v;
      } else {
        const float* fp = w + (long)oA * KF_ + g * 256 + ks * 32 + 16 * hi;
        #pragma unroll
        for (int e = 0; e < 8; ++e) {
          a[e]     = (_Float16)rintf(fp[e] * rs_w);
          a[e + 8] = (_Float16)rintf(fp[e + 8] * rs_w);
        }
      }

      // ---- B fragments from current LDS buffer ----
      const v16h* Bc = BsV[ks & 1];
      const int   nb = nhalf * 64;
      v16h b0 = Bc[(nb + 0  + col) * 2 + hi];
      v16h b1 = Bc[(nb + 16 + col) * 2 + hi];
      v16h b2 = Bc[(nb + 32 + col) * 2 + hi];
      v16h b3 = Bc[(nb + 48 + col) * 2 + hi];

      // ---- stage next K-step into the other buffer (overlaps WMMA) ----
      if (ks < 7) stageB(g * 256 + (ks + 1) * 32, BsV[(ks + 1) & 1]);

      acc0 = __builtin_amdgcn_wmma_f32_16x16x32_f16(false, a, false, b0,
                                                    (short)0, acc0, false, false);
      acc1 = __builtin_amdgcn_wmma_f32_16x16x32_f16(false, a, false, b1,
                                                    (short)0, acc1, false, false);
      acc2 = __builtin_amdgcn_wmma_f32_16x16x32_f16(false, a, false, b2,
                                                    (short)0, acc2, false, false);
      acc3 = __builtin_amdgcn_wmma_f32_16x16x32_f16(false, a, false, b3,
                                                    (short)0, acc3, false, false);

      if (ks < 7) __syncthreads();   // next buffer fully staged
    }

    // ---- per-subarray ADC quantize, accumulate over subarrays ----
    #pragma unroll
    for (int i = 0; i < 8; ++i) {
      tot0[i] += fminf(fmaxf(rintf(acc0[i] * invstep), -128.f), 127.f) * step;
      tot1[i] += fminf(fmaxf(rintf(acc1[i] * invstep), -128.f), 127.f) * step;
      tot2[i] += fminf(fmaxf(rintf(acc2[i] * invstep), -128.f), 127.f) * step;
      tot3[i] += fminf(fmaxf(rintf(acc3[i] * invstep), -128.f), 127.f) * step;
    }
  }

  // ---- store D: vgpr i -> m = i + 8*hi, lane col -> n ----
  const int l0 = lb * 128 + nhalf * 64;
  #pragma unroll
  for (int i = 0; i < 8; ++i) {
    const int  m = i + 8 * hi;
    const int  o = ob * 64 + ofrag * 16 + m;
    const long base = ((long)b * O_ + o) * (long)L_ + l0;
    out[base + 0  + col] = tot0[i];
    out[base + 16 + col] = tot1[i];
    out[base + 32 + col] = tot2[i];
    out[base + 48 + col] = tot3[i];
  }
}

// ---------------------------------------------------------------------------
// Host launcher
// ---------------------------------------------------------------------------
extern "C" void kernel_launch(void* const* d_in, const int* in_sizes, int n_in,
                              void* d_out, int out_size, void* d_ws, size_t ws_size,
                              hipStream_t stream) {
  const float* x = (const float*)d_in[0];   // [32,256,32,32]
  const float* w = (const float*)d_in[1];   // [256,256,3,3]
  float* out = (float*)d_out;               // [32,256,32,32] + a_loss

  const long nx = (long)B_ * C_ * HW_ * HW_;   // 8,388,608
  const long nw = (long)O_ * KF_;              // 589,824

  unsigned int* maxbuf = (unsigned int*)d_ws;
  const size_t off_w16  = 256;
  const size_t off_x16  = off_w16 + ((size_t)(nw * 2 + 255) / 256) * 256;
  const size_t need_x16 = off_x16 + (size_t)nx * 2;
  const int use_w16 = (ws_size >= off_x16) ? 1 : 0;
  const int use_x16 = (ws_size >= need_x16) ? 1 : 0;
  _Float16* W16 = (_Float16*)((char*)d_ws + off_w16);
  _Float16* X16 = (_Float16*)((char*)d_ws + off_x16);

  k_init<<<1, 64, 0, stream>>>(maxbuf, out, out_size);
  k_absmax<<<512, 256, 0, stream>>>(x, nx, maxbuf + 0);
  k_absmax<<<64, 256, 0, stream>>>(w, nw, maxbuf + 1);
  if (use_w16)
    k_quant<<<(int)((nw + 255) / 256), 256, 0, stream>>>(w, W16, nw, maxbuf, 1, 15.0f);
  if (use_x16)
    k_quant<<<(int)((nx + 255) / 256), 256, 0, stream>>>(x, X16, nx, maxbuf, 0, 255.0f);

  // 4 (o) * 8 (l) * 32 (b) = 1024 blocks
  if (use_x16 && use_w16)
    k_main<true,  true ><<<1024, 256, 0, stream>>>(x, w, X16, W16, maxbuf, out);
  else if (use_x16)
    k_main<true,  false><<<1024, 256, 0, stream>>>(x, w, X16, W16, maxbuf, out);
  else if (use_w16)
    k_main<false, true ><<<1024, 256, 0, stream>>>(x, w, X16, W16, maxbuf, out);
  else
    k_main<false, false><<<1024, 256, 0, stream>>>(x, w, X16, W16, maxbuf, out);
}
